// CausalPerformer_4913442586969
// MI455X (gfx1250) — compile-verified, matrix operand order checked by reading, not run on hardware
//
#include <hip/hip_runtime.h>

// ---------------------------------------------------------------------------
// Causal Performer attention, MI455X (gfx1250, wave32, WMMA).
//   - one-shot fp32->bf16 conversion pass (v_cvt_pk_bf16_f32)
//   - 4 big GEMMs (8192x2048x2048): bf16 V_WMMA_F32_16X16X32_BF16, fp32 accum
//       * ping-pong LDS tiles filled by async global->LDS DMA (ASYNCcnt)
//       * register double-buffered B fragments (ds_load overlaps wmma)
//   - feature map + causal linear-attention scan: small VALU kernels
// ---------------------------------------------------------------------------

typedef __attribute__((ext_vector_type(16))) __bf16 v16bf;
typedef __attribute__((ext_vector_type(8)))  __bf16 v8bf;
typedef __attribute__((ext_vector_type(8)))  float  v8f;
typedef int v4i_b __attribute__((vector_size(16)));   // matches async-LDS builtin param

#define D_MODEL 2048
#define N_HEADS 16
#define D_K     128
#define NB_FEAT 8
#define S_LEN   4096
#define B_SZ    2
#define EPS_F   1e-6f

#define TM 128     // block tile M
#define TN 128     // block tile N
#define TK 32      // K per WMMA step
#define LDST 40    // LDS row stride (bf16 elems): 80B, keeps 16B chunk alignment

#if __has_builtin(__builtin_amdgcn_global_load_async_to_lds_b128)
#define HAVE_ASYNC_LDS 1
#endif

struct __align__(16) U128 { unsigned int a, b, c, d; };

__device__ __forceinline__ void wait_async_zero() {
#if defined(HAVE_ASYNC_LDS)
#if __has_builtin(__builtin_amdgcn_s_wait_asynccnt)
    __builtin_amdgcn_s_wait_asynccnt(0);
#else
    asm volatile("s_wait_asynccnt 0x0" ::: "memory");
#endif
#endif
}

// fp32 -> bf16 (RNE), 8 elements per thread
__global__ __launch_bounds__(256)
void cvt_bf16_kernel(const float* __restrict__ src, unsigned short* __restrict__ dst,
                     int n8) {
    const int i = blockIdx.x * blockDim.x + threadIdx.x;
    if (i >= n8) return;
    const float4* s = (const float4*)(src + (size_t)i * 8);
    const float4 a = s[0], b = s[1];
    v8f t = { a.x, a.y, a.z, a.w, b.x, b.y, b.z, b.w };
    union { v8bf v; U128 u; } p;
    p.v = __builtin_convertvector(t, v8bf);               // v_cvt_pk_bf16_f32 x4
    *(U128*)(dst + (size_t)i * 8) = p.u;
}

// C[M,N] = A[M,K] @ W[N,K]^T   (A, W bf16 row-major, K-contiguous; C fp32)
__global__ __launch_bounds__(256)
void gemm_bf16_wmma(const unsigned short* __restrict__ A,
                    const unsigned short* __restrict__ W,
                    float* __restrict__ C, int M, int N, int K) {
    __shared__ __align__(16) unsigned short lA[2][TM * LDST];
    __shared__ __align__(16) unsigned short lB[2][TN * LDST];

    const int tid  = threadIdx.x;
    const int wave = tid >> 5;
    const int lane = tid & 31;
    const int m0   = blockIdx.x * TM;
    const int n0   = blockIdx.y * TN;

    const int r  = lane & 15;              // fragment row within 16x16 tile
    const int kb = (lane >> 4) * 8;        // K-chunk base per half-wave

    // cooperative-fill geometry: 2 chunks of 8 bf16 per matrix per thread
    int rows[2], cbs[2];
#pragma unroll
    for (int j = 0; j < 2; ++j) {
        const int c = tid + 256 * j;       // 512 chunks, 4 per 32-wide row
        rows[j] = c >> 2;
        cbs[j]  = (c & 3) * 8;
    }

    // issue LDS fill for one ping-pong buffer (async DMA if available)
    auto fill_tile = [&](int buf, int kk) {
#pragma unroll
        for (int j = 0; j < 2; ++j) {
            const unsigned short* ga = A + (size_t)(m0 + rows[j]) * K + kk + cbs[j];
            const unsigned short* gw = W + (size_t)(n0 + rows[j]) * K + kk + cbs[j];
            unsigned short* da = &lA[buf][rows[j] * LDST + cbs[j]];
            unsigned short* dw = &lB[buf][rows[j] * LDST + cbs[j]];
#if defined(HAVE_ASYNC_LDS)
            __builtin_amdgcn_global_load_async_to_lds_b128(
                (v4i_b*)ga, (v4i_b*)da, 0, 0);
            __builtin_amdgcn_global_load_async_to_lds_b128(
                (v4i_b*)gw, (v4i_b*)dw, 0, 0);
#else
            *(U128*)da = *(const U128*)ga;
            *(U128*)dw = *(const U128*)gw;
#endif
        }
    };

    v8f acc[8] = {};                       // 8 N-tiles of 16x16 per wave

    fill_tile(0, 0);
    wait_async_zero();
    __syncthreads();

    const int nsteps = K / TK;
    for (int step = 0; step < nsteps; ++step) {
        const int cur = step & 1;

        // DMA next tile into the other buffer while we compute on this one
        if (step + 1 < nsteps) fill_tile(cur ^ 1, (step + 1) * TK);

        union FragU { v16bf v; U128 q[2]; };
        FragU a;
        {
            const unsigned short* pA = &lA[cur][(wave * 16 + r) * LDST + kb];
            a.q[0] = *(const U128*)pA;
            a.q[1] = *(const U128*)(pA + 16);
        }
        // register double-buffered B fragments
        FragU bc, bn;
        {
            const unsigned short* pB = &lB[cur][r * LDST + kb];
            bc.q[0] = *(const U128*)pB;
            bc.q[1] = *(const U128*)(pB + 16);
        }
#pragma unroll
        for (int nt = 0; nt < 8; ++nt) {
            if (nt < 7) {
                const unsigned short* pBn = &lB[cur][((nt + 1) * 16 + r) * LDST + kb];
                bn.q[0] = *(const U128*)pBn;
                bn.q[1] = *(const U128*)(pBn + 16);
            }
            acc[nt] = __builtin_amdgcn_wmma_f32_16x16x32_bf16(
                false, a.v, false, bc.v, (short)0, acc[nt], false, false);
            bc = bn;
        }

        wait_async_zero();                 // next buffer fully resident
        __syncthreads();
    }

    // ---- store: VGPR j -> M = j + 8*(lane>>4), N = lane&15 ----
    const int orow = m0 + wave * 16 + 8 * (lane >> 4);
    const int ocol = n0 + (lane & 15);
#pragma unroll
    for (int nt = 0; nt < 8; ++nt) {
#pragma unroll
        for (int j = 0; j < 8; ++j) {
            C[(size_t)(orow + j) * N + ocol + nt * 16] = acc[nt][j];
        }
    }
}

// x' = normalize( exp(-0.5*(omega . x)^2) ) over NB_FEAT, per (b,s,h)
__global__ __launch_bounds__(256)
void feature_kernel(const float* __restrict__ xh, const float* __restrict__ omega,
                    float* __restrict__ xp, int total) {
    __shared__ float lom[NB_FEAT * D_K];
    for (int i = threadIdx.x; i < NB_FEAT * D_K; i += blockDim.x) lom[i] = omega[i];
    __syncthreads();

    const int i = blockIdx.x * blockDim.x + threadIdx.x;
    if (i >= total) return;
    const int h  = i & (N_HEADS - 1);
    const int bs = i >> 4;

    const float4* xr = (const float4*)(xh + (size_t)bs * D_MODEL + h * D_K);
    float dot[NB_FEAT] = {};
    for (int d4 = 0; d4 < D_K / 4; ++d4) {
        const float4 x = xr[d4];
#pragma unroll
        for (int f = 0; f < NB_FEAT; ++f) {
            const float4 o = ((const float4*)(lom + f * D_K))[d4];
            dot[f] += x.x * o.x + x.y * o.y + x.z * o.z + x.w * o.w;
        }
    }
    float phi[NB_FEAT], s = 0.f;
#pragma unroll
    for (int f = 0; f < NB_FEAT; ++f) {
        phi[f] = __expf(-0.5f * dot[f] * dot[f]);
        s += phi[f];
    }
    const float inv = 1.f / (s + EPS_F);
    float4 p0 = { phi[0] * inv, phi[1] * inv, phi[2] * inv, phi[3] * inv };
    float4 p1 = { phi[4] * inv, phi[5] * inv, phi[6] * inv, phi[7] * inv };
    float4* op = (float4*)(xp + (size_t)i * NB_FEAT);
    op[0] = p0; op[1] = p1;
}

// causal linear-attention scan: one block per (b,h); lane d owns column d of
// the 8x128 running state.  out = (q'.KV_cum)/(q'.K_cum + eps)
__global__ __launch_bounds__(128)
void scan_kernel(const float* __restrict__ qp, const float* __restrict__ kp,
                 const float* __restrict__ vh, float* __restrict__ attn) {
    const int b = blockIdx.x >> 4;          // / N_HEADS
    const int h = blockIdx.x & (N_HEADS - 1);
    const int d = threadIdx.x;

    float st[NB_FEAT] = {};                 // KV state column
    float ks[NB_FEAT] = {};                 // K cumulative

    for (int t = 0; t < S_LEN; ++t) {
        const size_t bs = (size_t)b * S_LEN + t;
        const size_t pb = (bs * N_HEADS + h) * NB_FEAT;
        const float4 k0 = ((const float4*)(kp + pb))[0];
        const float4 k1 = ((const float4*)(kp + pb))[1];
        const float4 q0 = ((const float4*)(qp + pb))[0];
        const float4 q1 = ((const float4*)(qp + pb))[1];
        const float  v  = vh[bs * D_MODEL + h * D_K + d];

        const float kf[8] = { k0.x, k0.y, k0.z, k0.w, k1.x, k1.y, k1.z, k1.w };
        const float qf[8] = { q0.x, q0.y, q0.z, q0.w, q1.x, q1.y, q1.z, q1.w };

        float num = 0.f, den = 0.f;
#pragma unroll
        for (int f = 0; f < NB_FEAT; ++f) {
            ks[f] += kf[f];
            st[f] += kf[f] * v;
            num   += qf[f] * st[f];
            den   += qf[f] * ks[f];
        }
        attn[bs * D_MODEL + h * D_K + d] = num / (den + EPS_F);
    }
}

extern "C" void kernel_launch(void* const* d_in, const int* in_sizes, int n_in,
                              void* d_out, int out_size, void* d_ws, size_t ws_size,
                              hipStream_t stream) {
    const float* q     = (const float*)d_in[0];
    const float* k     = (const float*)d_in[1];
    const float* v     = (const float*)d_in[2];
    const float* w_q   = (const float*)d_in[3];
    const float* w_k   = (const float*)d_in[4];
    const float* w_v   = (const float*)d_in[5];
    const float* w_o   = (const float*)d_in[6];
    const float* omega = (const float*)d_in[7];

    const int M = B_SZ * S_LEN;       // 8192
    const int N = D_MODEL, K = D_MODEL;
    const size_t NEL = (size_t)M * D_MODEL;                 // 16,777,216
    const size_t WEL = (size_t)D_MODEL * D_MODEL;           //  4,194,304
    const size_t PEL = (size_t)M * N_HEADS * NB_FEAT;       //  1,048,576

    // ---- workspace layout ----
    // bf16: qb kb vb | wqb wkb wvb wob      (as ushort)
    // fp32: qh kh vh | q' k'                (attn reuses qh; attn_bf16 reuses qb)
    unsigned short* qb  = (unsigned short*)d_ws;
    unsigned short* kb  = qb + NEL;
    unsigned short* vb  = kb + NEL;
    unsigned short* wqb = vb + NEL;
    unsigned short* wkb = wqb + WEL;
    unsigned short* wvb = wkb + WEL;
    unsigned short* wob = wvb + WEL;
    float* qh   = (float*)(wob + WEL);
    float* kh   = qh + NEL;
    float* vh   = kh + NEL;
    float* qpb  = vh + NEL;
    float* kpb  = qpb + PEL;
    float* attn = qh;                 // qh dead after feature_kernel
    unsigned short* attnb = qb;       // qb dead after first GEMM

    // ---- fp32 -> bf16 pre-pass ----
    const int an8 = (int)(NEL / 8), wn8 = (int)(WEL / 8);
    cvt_bf16_kernel<<<(an8 + 255) / 256, 256, 0, stream>>>(q, qb, an8);
    cvt_bf16_kernel<<<(an8 + 255) / 256, 256, 0, stream>>>(k, kb, an8);
    cvt_bf16_kernel<<<(an8 + 255) / 256, 256, 0, stream>>>(v, vb, an8);
    cvt_bf16_kernel<<<(wn8 + 255) / 256, 256, 0, stream>>>(w_q, wqb, wn8);
    cvt_bf16_kernel<<<(wn8 + 255) / 256, 256, 0, stream>>>(w_k, wkb, wn8);
    cvt_bf16_kernel<<<(wn8 + 255) / 256, 256, 0, stream>>>(w_v, wvb, wn8);
    cvt_bf16_kernel<<<(wn8 + 255) / 256, 256, 0, stream>>>(w_o, wob, wn8);

    // ---- projections ----
    const dim3 gg(M / TM, N / TN);    // 64 x 16 blocks
    gemm_bf16_wmma<<<gg, 256, 0, stream>>>(qb, wqb, qh, M, N, K);
    gemm_bf16_wmma<<<gg, 256, 0, stream>>>(kb, wkb, kh, M, N, K);
    gemm_bf16_wmma<<<gg, 256, 0, stream>>>(vb, wvb, vh, M, N, K);

    // ---- feature maps ----
    const int total = M * N_HEADS;    // 131072
    feature_kernel<<<(total + 255) / 256, 256, 0, stream>>>(qh, omega, qpb, total);
    feature_kernel<<<(total + 255) / 256, 256, 0, stream>>>(kh, omega, kpb, total);

    // ---- causal linear-attention scan ----
    scan_kernel<<<B_SZ * N_HEADS, 128, 0, stream>>>(qpb, kpb, vh, attn);

    // ---- output projection ----
    cvt_bf16_kernel<<<(an8 + 255) / 256, 256, 0, stream>>>(attn, attnb, an8);
    gemm_bf16_wmma<<<gg, 256, 0, stream>>>(attnb, wob, (float*)d_out, M, N, K);
}
